// Attention_56178172231843
// MI455X (gfx1250) — matrix operand order repaired and checked by reading, not compiled, not run on hardware
//
#include <hip/hip_runtime.h>

typedef __attribute__((ext_vector_type(16))) __bf16 v16bf;
typedef __attribute__((ext_vector_type(8)))  float  v8f;
typedef __attribute__((ext_vector_type(4)))  unsigned u32x4;
typedef __attribute__((ext_vector_type(8)))  unsigned u32x8;

struct alignas(16) B16 { unsigned long long lo, hi; };
union Frag { v16bf v; B16 u[2]; };

__device__ __forceinline__ float bf2f(unsigned short h) {
  return __uint_as_float(((unsigned)h) << 16);
}
__device__ __forceinline__ unsigned short f2bf(float f) {
  unsigned u = __float_as_uint(f);
  return (unsigned short)((u + 0x7FFFu + ((u >> 16) & 1u)) >> 16);
}

// ---------------- conversions ----------------
__global__ __launch_bounds__(256) void cvt_f32_bf16(const float* __restrict__ in,
                                                    unsigned short* __restrict__ out, int n) {
  int i = blockIdx.x * 256 + threadIdx.x;
  if (i < n) out[i] = f2bf(in[i]);
}

// w_qkv [768 x 2304] -> transposed [2304 x 768], q-scale (0.125) folded into rows < 768
__global__ __launch_bounds__(256) void cvt_wqkv_t(const float* __restrict__ in,
                                                  unsigned short* __restrict__ out) {
  int i = blockIdx.x * 256 + threadIdx.x;  // 768*2304 exact
  int col = i % 2304;
  int row = i / 2304;
  float f = in[i];
  if (col < 768) f *= 0.125f;
  out[col * 768 + row] = f2bf(f);
}

// w_out [768 x 768] -> transposed
__global__ __launch_bounds__(256) void cvt_wout_t(const float* __restrict__ in,
                                                  unsigned short* __restrict__ out) {
  int i = blockIdx.x * 256 + threadIdx.x;  // 768*768 exact
  int col = i % 768;
  int row = i / 768;
  out[col * 768 + row] = f2bf(in[i]);
}

// qkv [8192 x 2304] -> q/k planes [b,h,1024,64]; v stored transposed [b,h,64,1024]
__global__ __launch_bounds__(256) void reshape_qkv(const unsigned short* __restrict__ qkv,
                                                   unsigned short* __restrict__ qb,
                                                   unsigned short* __restrict__ kb,
                                                   unsigned short* __restrict__ vtb) {
  size_t i = (size_t)blockIdx.x * 256 + threadIdx.x;  // 8192*2304 exact
  int col = (int)(i % 2304);
  size_t row = i / 2304;
  int sel = col / 768, c = col % 768;
  int h = c >> 6, d = c & 63;
  size_t b = row >> 10, ii = row & 1023;
  unsigned short val = qkv[i];
  if (sel == 2) {
    vtb[((((b * 12 + h) << 6) + d) << 10) + ii] = val;  // [b,h,d,j]
  } else {
    size_t dst = ((((b * 12 + h) << 10) + ii) << 6) + d;
    (sel == 0 ? qb : kb)[dst] = val;
  }
}

// ---------------- TDM tile load: [rows=64 x TK] bf16, padded LDS rows ----------------
#define TM 64
#define TN 64
#define TK 32
#define KS 48  // LDS halfwords per row = 32 data + 16 pad (96B, keeps b128 alignment)

__device__ __forceinline__ void tdm_load_tile(const void* gptr, void* ldsPtr, int ldaElems) {
  unsigned long long ga = (unsigned long long)gptr;
  unsigned lds = (unsigned)(unsigned long long)ldsPtr;  // low 32 bits = LDS byte offset
  u32x4 g0;
  g0.s0 = 1u;                                             // count=1, user mode
  g0.s1 = lds;                                            // lds_addr
  g0.s2 = (unsigned)ga;                                   // global_addr[31:0]
  g0.s3 = ((unsigned)(ga >> 32) & 0x01FFFFFFu) | 0x80000000u;  // addr[56:32], type=2
  u32x8 g1;
  // data_size=1(2B) | pad_enable | pad_interval=3(16 DW) | pad_amount=7(8 DW)
  g1.s0 = (1u << 16) | (1u << 20) | (3u << 22) | (7u << 25);
  g1.s1 = ((unsigned)ldaElems & 0xFFFFu) << 16;           // tensor_dim0[15:0]
  g1.s2 = (((unsigned)ldaElems >> 16) & 0xFFFFu) | (0xFFFFu << 16);  // dim0 hi | dim1 lo
  g1.s3 = 0u | ((unsigned)TK << 16);                      // dim1 hi | tile_dim0=32
  g1.s4 = 64u;                                            // tile_dim1=64, tile_dim2=0
  g1.s5 = (unsigned)ldaElems;                             // tensor_dim0_stride[31:0]
  g1.s6 = 0u;                                             // stride hi | dim1_stride lo
  g1.s7 = 0u;
  asm volatile("tensor_load_to_lds %0, %1" :: "s"(g0), "s"(g1) : "memory");
}

// ---------------- generic bf16 WMMA GEMM, B is [N x K] row-major ----------------
template <int EPI>  // 0: bf16 store, 1: f32 + bias store
__global__ __launch_bounds__(128) void gemm_wmma_bf16(
    const unsigned short* __restrict__ A, long sAz, int lda,
    const unsigned short* __restrict__ B, long sBz, int ldb,
    void* __restrict__ Cv, long sC0, long sC1, int divH, int ldc,
    const float* __restrict__ bias, int K) {
  __shared__ unsigned short lA[2][TM * KS];
  __shared__ unsigned short lB[2][TN * KS];
  const int tid = threadIdx.x;
  const int lane = tid & 31;
  const int wid = tid >> 5;
  const int wm = (wid >> 1) * 32;
  const int wn = (wid & 1) * 32;
  const int z = blockIdx.z;
  const int m0 = blockIdx.x * TM;
  const int n0 = blockIdx.y * TN;
  A += (long)z * sAz + (long)m0 * lda;
  B += (long)z * sBz + (long)n0 * ldb;
  const long cofs = (long)(z / divH) * sC0 + (long)(z % divH) * sC1;

  v8f acc[2][2] = {};
  const int kh = lane >> 4;
  const int lm = lane & 15;
  const int nk = K / TK;

  if (wid == 0) {  // prime buffer 0 via Tensor Data Mover (EXEC-independent DMA)
    tdm_load_tile(A, &lA[0][0], lda);
    tdm_load_tile(B, &lB[0][0], ldb);
  }

  for (int it = 0; it < nk; ++it) {
    const int cur = it & 1;
    if (wid == 0) {
      if (it + 1 < nk) {  // issue next tile into other buffer, then drain current
        tdm_load_tile(A + (it + 1) * TK, &lA[cur ^ 1][0], lda);
        tdm_load_tile(B + (it + 1) * TK, &lB[cur ^ 1][0], ldb);
        __builtin_amdgcn_s_wait_tensorcnt(2);
      } else {
        __builtin_amdgcn_s_wait_tensorcnt(0);
      }
    }
    __syncthreads();  // current tile resident in LDS for all waves

    // fragments per CDNA5 16-bit WMMA layouts (ISA 7.12.2)
    const unsigned short* lAc = &lA[cur][0];
    const unsigned short* lBc = &lB[cur][0];
    Frag af[2], bfr[2];
#pragma unroll
    for (int mi = 0; mi < 2; ++mi) {
      // A 16x32: lanes0-15 K{0..7,16..23}, lanes16-31 K{8..15,24..31}
      const unsigned short* ap = &lAc[(wm + mi * 16 + lm) * KS + kh * 8];
      af[mi].u[0] = *(const B16*)ap;
      af[mi].u[1] = *(const B16*)(ap + 16);
    }
#pragma unroll
    for (int ni = 0; ni < 2; ++ni) {
      // B 32x16: lanes0-15 K{0..7,16..23}, lanes16-31 K{8..15,24..31} (lB rows are N)
      const unsigned short* bp = &lBc[(wn + ni * 16 + lm) * KS + kh * 8];
      bfr[ni].u[0] = *(const B16*)bp;
      bfr[ni].u[1] = *(const B16*)(bp + 16);
    }
#pragma unroll
    for (int mi = 0; mi < 2; ++mi)
#pragma unroll
      for (int ni = 0; ni < 2; ++ni)
        acc[mi][ni] = __builtin_amdgcn_wmma_f32_16x16x32_bf16(
            false, af[mi].v, false, bfr[ni].v, (short)0, acc[mi][ni], false, false);
    __syncthreads();  // all reads done before TDM overwrites this buffer
  }

  // C layout: VGPR e -> M = e + (lane>=16 ? 8:0), N = lane&15
#pragma unroll
  for (int mi = 0; mi < 2; ++mi)
#pragma unroll
    for (int ni = 0; ni < 2; ++ni) {
      const int mBase = m0 + wm + mi * 16 + (kh ? 8 : 0);
      const int col = n0 + wn + ni * 16 + lm;
#pragma unroll
      for (int e = 0; e < 8; ++e) {
        const long idx = cofs + (long)(mBase + e) * ldc + col;
        const float val = acc[mi][ni][e];
        if (EPI == 0)
          ((unsigned short*)Cv)[idx] = f2bf(val);
        else
          ((float*)Cv)[idx] = val + bias[col];
      }
    }
}

// ---------------- softmax over rows of 1024 (in place, bf16) ----------------
__global__ __launch_bounds__(128) void softmax_rows(unsigned short* __restrict__ s) {
  const int row = blockIdx.x * 4 + (threadIdx.x >> 5);
  const int lane = threadIdx.x & 31;
  unsigned short* p = s + (size_t)row * 1024 + lane * 32;
  B16 raw[4];
#pragma unroll
  for (int c = 0; c < 4; ++c) raw[c] = ((const B16*)p)[c];
  const unsigned short* e = (const unsigned short*)raw;
  float v[32];
  float mx = -3.0e38f;
#pragma unroll
  for (int c = 0; c < 32; ++c) {
    v[c] = bf2f(e[c]);
    mx = fmaxf(mx, v[c]);
  }
#pragma unroll
  for (int o = 16; o > 0; o >>= 1) mx = fmaxf(mx, __shfl_xor(mx, o, 32));
  float sum = 0.f;
#pragma unroll
  for (int c = 0; c < 32; ++c) {
    v[c] = __expf(v[c] - mx);
    sum += v[c];
  }
#pragma unroll
  for (int o = 16; o > 0; o >>= 1) sum += __shfl_xor(sum, o, 32);
  const float inv = 1.0f / sum;
  unsigned short ov[32];
#pragma unroll
  for (int c = 0; c < 32; ++c) ov[c] = f2bf(v[c] * inv);
#pragma unroll
  for (int c = 0; c < 4; ++c) ((B16*)p)[c] = ((const B16*)ov)[c];
}

// ---------------- re-attention head-mix + LayerNorm over heads (in place) ----------------
__global__ __launch_bounds__(256) void reattn_mix_ln(unsigned short* __restrict__ s,
                                                     const float* __restrict__ W,
                                                     const float* __restrict__ gam,
                                                     const float* __restrict__ bet) {
  __shared__ float wS[144], gS[12], bS[12];
  if (threadIdx.x < 144) wS[threadIdx.x] = W[threadIdx.x];
  if (threadIdx.x < 12) {
    gS[threadIdx.x] = gam[threadIdx.x];
    bS[threadIdx.x] = bet[threadIdx.x];
  }
  __syncthreads();
  size_t idx = (size_t)blockIdx.x * 256 + threadIdx.x;  // 8*1024*1024 exact
  size_t b = idx >> 20;
  size_t ij = idx & 1048575;
  unsigned short* base = s + ((b * 12) << 20) + ij;
  float a[12];
#pragma unroll
  for (int h = 0; h < 12; ++h) a[h] = bf2f(base[(size_t)h << 20]);
  float m[12];
  float mean = 0.f;
#pragma unroll
  for (int g = 0; g < 12; ++g) {
    float accv = 0.f;
#pragma unroll
    for (int h = 0; h < 12; ++h) accv += a[h] * wS[h * 12 + g];
    m[g] = accv;
    mean += accv;
  }
  mean *= (1.f / 12.f);
  float var = 0.f;
#pragma unroll
  for (int g = 0; g < 12; ++g) {
    float d = m[g] - mean;
    var += d * d;
  }
  var *= (1.f / 12.f);
  const float rs = rsqrtf(var + 1e-6f);
#pragma unroll
  for (int g = 0; g < 12; ++g)
    base[(size_t)g << 20] = f2bf((m[g] - mean) * rs * gS[g] + bS[g]);
}

// ---------------- launcher ----------------
extern "C" void kernel_launch(void* const* d_in, const int* in_sizes, int n_in,
                              void* d_out, int out_size, void* d_ws, size_t ws_size,
                              hipStream_t stream) {
  const float* x = (const float*)d_in[0];       // [8,1024,768]
  const float* w_qkv = (const float*)d_in[1];   // [768,2304]
  const float* reattn = (const float*)d_in[2];  // [12,12]
  const float* ln_s = (const float*)d_in[3];    // [12]
  const float* ln_b = (const float*)d_in[4];    // [12]
  const float* w_out = (const float*)d_in[5];   // [768,768]
  const float* b_out = (const float*)d_in[6];   // [768]
  float* out = (float*)d_out;                   // [8,1024,768]

  char* ws = (char*)d_ws;
  size_t ofs = 0;
  auto alloc = [&](size_t elems) {
    size_t cur = ofs;
    ofs += (elems * 2 + 255) & ~(size_t)255;
    return (unsigned short*)(ws + cur);
  };
  unsigned short* xb     = alloc((size_t)8192 * 768);
  unsigned short* wqkvT  = alloc((size_t)2304 * 768);
  unsigned short* woutT  = alloc((size_t)768 * 768);
  unsigned short* qkvb   = alloc((size_t)8192 * 2304);
  unsigned short* qb     = alloc((size_t)96 * 1024 * 64);
  unsigned short* kb     = alloc((size_t)96 * 1024 * 64);
  unsigned short* vtb    = alloc((size_t)96 * 64 * 1024);
  unsigned short* sc     = alloc((size_t)96 * 1024 * 1024);  // attention matrix, bf16
  unsigned short* avo    = alloc((size_t)8192 * 768);

  // 1) converts (+ weight transposes so every GEMM consumes B as [N x K])
  cvt_f32_bf16<<<24576, 256, 0, stream>>>(x, xb, 8192 * 768);
  cvt_wqkv_t<<<6912, 256, 0, stream>>>(w_qkv, wqkvT);
  cvt_wout_t<<<2304, 256, 0, stream>>>(w_out, woutT);

  // 2) qkv = x @ w_qkv  (8192 x 2304 x 768)
  gemm_wmma_bf16<0><<<dim3(128, 36, 1), 128, 0, stream>>>(
      xb, 0L, 768, wqkvT, 0L, 768, qkvb, 0L, 0L, 1, 2304, nullptr, 768);

  // 3) split to per-head planes (v transposed for the AV GEMM)
  reshape_qkv<<<73728, 256, 0, stream>>>(qkvb, qb, kb, vtb);

  // 4) scores[b,h] = q @ k^T   (96 batches of 1024 x 1024 x 64)
  gemm_wmma_bf16<0><<<dim3(16, 16, 96), 128, 0, stream>>>(
      qb, 65536L, 64, kb, 65536L, 64, sc, 1048576L, 0L, 1, 1024, nullptr, 64);

  // 5) softmax over j (98304 rows, 4 rows per 128-thread block)
  softmax_rows<<<24576, 128, 0, stream>>>(sc);

  // 6) re-attention mix across heads + LayerNorm over heads, in place
  reattn_mix_ln<<<32768, 256, 0, stream>>>(sc, reattn, ln_s, ln_b);

  // 7) out[b,:,g*64+d] = attn[b,g] @ v[b,g]   (96 batches of 1024 x 64 x 1024)
  gemm_wmma_bf16<0><<<dim3(16, 1, 96), 128, 0, stream>>>(
      sc, 1048576L, 1024, vtb, 65536L, 1024, avo,
      (long)1024 * 768, 64L, 12, 768, nullptr, 1024);

  // 8) final projection: out = avo @ w_out + b_out  (8192 x 768 x 768), f32
  gemm_wmma_bf16<1><<<dim3(128, 12, 1), 128, 0, stream>>>(
      avo, 0L, 768, woutT, 0L, 768, out, 0L, 0L, 1, 768, b_out, 768);
}
